// MultiHeadSelfAttention_35356170780686
// MI455X (gfx1250) — matrix operand order repaired and checked by reading, not compiled, not run on hardware
//
#include <hip/hip_runtime.h>
#include <hip/hip_bf16.h>
#include <cstdint>
#include <cstddef>

#define BATCH   8
#define SEQ     2048
#define DMODEL  1024
#define MROWS   (BATCH * SEQ)   // 16384 flattened (b, s) rows

typedef __attribute__((ext_vector_type(16))) __bf16 v16bf;
typedef __attribute__((ext_vector_type(8)))  float  v8f;

union FragBF {
    v16bf v;
    uint4 q[2];
};

__device__ __forceinline__ unsigned short f32_to_bf16(float f) {
    unsigned int u = __builtin_bit_cast(unsigned int, f);
    u = u + 0x7FFFu + ((u >> 16) & 1u);   // round-to-nearest-even
    return (unsigned short)(u >> 16);
}

__device__ __forceinline__ v8f wmma_bf16(const FragBF& a, const FragBF& b, v8f c) {
    return __builtin_amdgcn_wmma_f32_16x16x32_bf16(
        /*neg_a=*/false, a.v, /*neg_b=*/false, b.v,
        /*c_mod=*/(short)0, c, /*reuse_a=*/false, /*reuse_b=*/false);
}

// ---------------------------------------------------------------------------
// fp32 -> bf16 conversion (grid-stride)
// ---------------------------------------------------------------------------
__global__ void cvt_f32_to_bf16(const float* __restrict__ src,
                                unsigned short* __restrict__ dst, int n) {
    int i = blockIdx.x * blockDim.x + threadIdx.x;
    int stride = gridDim.x * blockDim.x;
    for (; i < n; i += stride) dst[i] = f32_to_bf16(src[i]);
}

// ---------------------------------------------------------------------------
// C[MROWS, DMODEL] = A[MROWS, DMODEL] @ W[DMODEL, DMODEL]^T   (bf16 -> WMMA)
// W stored (out, in) row-major, so WMMA B-fragment (lane = out col, K contig)
// loads contiguous rows of W; A-fragment loads contiguous rows of A.
// MODE 0: store bf16 row-major.  MODE 1: store bf16 transposed per batch
// (Vt[b, d, s]).  MODE 2: store fp32 row-major.
// Block: 256 threads = 8 waves in 4(M) x 2(N); wave tile 64x64
// -> 16 WMMAs per 16 b128 loads per k-step (1:1 tensor:vmem issue).
// ---------------------------------------------------------------------------
template <int MODE>
__global__ void gemm_bf16_wmma(const unsigned short* __restrict__ A,
                               const unsigned short* __restrict__ W,
                               void* __restrict__ Cout) {
    const int lane  = threadIdx.x & 31;
    const int wave  = threadIdx.x >> 5;
    const int wm    = wave >> 1;          // 0..3
    const int wn    = wave & 1;           // 0..1
    const int m_base = blockIdx.y * 256 + wm * 64;
    const int n_base = blockIdx.x * 128 + wn * 64;
    const int lhalf = lane >> 4;          // 0 or 1
    const int l15   = lane & 15;

    v8f acc[4][4];
#pragma unroll
    for (int mi = 0; mi < 4; ++mi)
#pragma unroll
        for (int ni = 0; ni < 4; ++ni)
            acc[mi][ni] = (v8f){0.f, 0.f, 0.f, 0.f, 0.f, 0.f, 0.f, 0.f};

    const unsigned short* arow = A + (size_t)(m_base + l15) * DMODEL + lhalf * 16;
    const unsigned short* brow = W + (size_t)(n_base + l15) * DMODEL + lhalf * 16;

#pragma unroll 1
    for (int k0 = 0; k0 < DMODEL; k0 += 32) {
        FragBF a[4], b[4];
#pragma unroll
        for (int mi = 0; mi < 4; ++mi) {
            // A frag: lane half 0 -> K 0..15 ; half 1 -> K 16..31 (contig 32B)
            const unsigned short* p = arow + (size_t)mi * 16 * DMODEL + k0;
            a[mi].q[0] = *(const uint4*)p;
            a[mi].q[1] = *(const uint4*)(p + 8);
        }
#pragma unroll
        for (int ni = 0; ni < 4; ++ni) {
            // B frag: lane half 0 -> K 0..15 ; half 1 -> K 16..31 (contig 32B)
            const unsigned short* p = brow + (size_t)ni * 16 * DMODEL + k0;
            b[ni].q[0] = *(const uint4*)p;
            b[ni].q[1] = *(const uint4*)(p + 8);
        }
        if (k0 + 32 < DMODEL) {
            __builtin_prefetch(arow + k0 + 32, 0, 3);   // global_prefetch_b8
            __builtin_prefetch(brow + k0 + 32, 0, 3);
        }
#pragma unroll
        for (int mi = 0; mi < 4; ++mi)
#pragma unroll
            for (int ni = 0; ni < 4; ++ni)
                acc[mi][ni] = wmma_bf16(a[mi], b[ni], acc[mi][ni]);
    }

#pragma unroll
    for (int mi = 0; mi < 4; ++mi)
#pragma unroll
        for (int ni = 0; ni < 4; ++ni)
#pragma unroll
            for (int i = 0; i < 8; ++i) {
                const int row = m_base + mi * 16 + i + lhalf * 8;
                const int col = n_base + ni * 16 + l15;
                const float val = acc[mi][ni][i];
                if (MODE == 0) {
                    ((unsigned short*)Cout)[(size_t)row * DMODEL + col] =
                        f32_to_bf16(val);
                } else if (MODE == 1) {
                    const int bb = row >> 11;            // / SEQ
                    const int ss = row & (SEQ - 1);
                    ((unsigned short*)Cout)[((size_t)bb * DMODEL + col) * SEQ + ss] =
                        f32_to_bf16(val);
                } else {
                    ((float*)Cout)[(size_t)row * DMODEL + col] = val;
                }
            }
}

// ---------------------------------------------------------------------------
// Causal attention for one (batch, 16-query tile) per workgroup.
// Dynamic LDS: scores fp32 [16][SEQ] | probs bf16 [16][SEQ] | Q bf16 [16][D]
//              | reductions.  ~226 KB (exploits the 320 KB WGP LDS).
// 8 waves: each wave computes a 16x64 key strip of scores (1 shared A-frag
// feeds 4 WMMAs per d-step); P@V splits d_v 128-wide per wave.
// ---------------------------------------------------------------------------
__global__ void flash_attn(const unsigned short* __restrict__ Qb,
                           const unsigned short* __restrict__ Kb,
                           const unsigned short* __restrict__ Vt,   // [B, D, S]
                           unsigned short* __restrict__ Ob) {
    extern __shared__ char smem[];
    float*          sS   = (float*)smem;                                   // 16*SEQ f32
    unsigned short* sP   = (unsigned short*)(smem + 16 * SEQ * 4);         // 16*SEQ bf16
    unsigned short* sQ   = (unsigned short*)(smem + 16 * SEQ * 6);         // 16*DMODEL bf16
    float*          sRed = (float*)(smem + 16 * SEQ * 6 + 16 * DMODEL * 2);
    float*          sMax = sRed + 256;
    float*          sInv = sMax + 16;

    const int tid   = threadIdx.x;
    const int lane  = tid & 31;
    const int wave  = tid >> 5;
    const int lhalf = lane >> 4;
    const int l15   = lane & 15;
    const int qt    = blockIdx.x;          // query tile index
    const int b     = blockIdx.y;
    const int q0    = qt * 16;
    const int limit = q0 + 16;             // causal key bound for this tile
    const float scale = 0.03125f;          // 1/sqrt(1024)

    // ---- stage Q tile (16 x DMODEL bf16) into LDS ----
    for (int idx = tid; idx < 16 * DMODEL / 8; idx += 256) {
        const int e = idx * 8;
        const int r = e >> 10;             // / DMODEL
        const int c = e & (DMODEL - 1);
        *(uint4*)(sQ + r * DMODEL + c) =
            *(const uint4*)(Qb + ((size_t)(b * SEQ + q0 + r)) * DMODEL + c);
    }
    __syncthreads();

    // ---- scores = scale * Q K^T, causal-masked, into LDS (fp32) ----
    // Each wave owns 64-key strips; keys past `limit` are valid memory
    // (< SEQ) and are masked to -inf at the store, so the hot loop has no
    // branches: per d-step = 2 ds_load_b128 + 8 global_load_b128 + 4 WMMA.
#pragma unroll 1
    for (int strip = wave; strip * 64 < limit; strip += 8) {
        const int kbase = strip * 64;
        v8f acc[4];
#pragma unroll
        for (int t = 0; t < 4; ++t)
            acc[t] = (v8f){0.f, 0.f, 0.f, 0.f, 0.f, 0.f, 0.f, 0.f};
        const unsigned short* krow =
            Kb + ((size_t)(b * SEQ + kbase + l15)) * DMODEL + lhalf * 16;
#pragma unroll 1
        for (int d0 = 0; d0 < DMODEL; d0 += 32) {
            FragBF a;
            const unsigned short* pa = sQ + l15 * DMODEL + d0 + lhalf * 16;
            a.q[0] = *(const uint4*)pa;
            a.q[1] = *(const uint4*)(pa + 8);
#pragma unroll
            for (int t = 0; t < 4; ++t) {
                FragBF kb;
                const unsigned short* pb = krow + (size_t)t * 16 * DMODEL + d0;
                kb.q[0] = *(const uint4*)pb;
                kb.q[1] = *(const uint4*)(pb + 8);
                acc[t] = wmma_bf16(a, kb, acc[t]);
            }
        }
#pragma unroll
        for (int t = 0; t < 4; ++t)
#pragma unroll
            for (int i = 0; i < 8; ++i) {
                const int r = i + lhalf * 8;               // 0..15 within tile
                const int c = kbase + t * 16 + l15;        // global key
                const float v = (c <= q0 + r) ? acc[t][i] * scale : -3.0e38f;
                sS[r * SEQ + c] = v;
            }
    }
    __syncthreads();

    // ---- max-shift softmax over keys [0, limit) : 16 threads per row ----
    {
        const int r = tid >> 4;
        const int sub = tid & 15;
        float m = -3.0e38f;
        for (int c = sub; c < limit; c += 16) m = fmaxf(m, sS[r * SEQ + c]);
        sRed[r * 16 + sub] = m;
        __syncthreads();
        if (tid < 16) {
            float mm = sRed[tid * 16];
            for (int j = 1; j < 16; ++j) mm = fmaxf(mm, sRed[tid * 16 + j]);
            sMax[tid] = mm;
        }
        __syncthreads();
        const float rm = sMax[r];
        float ssum = 0.f;
        for (int c = sub; c < limit; c += 16) {
            const float e = __expf(sS[r * SEQ + c] - rm);
            sP[r * SEQ + c] = f32_to_bf16(e);
            ssum += e;
        }
        sRed[r * 16 + sub] = ssum;
        // zero-pad P to a multiple of 32 keys for the WMMA K-loop
        if (limit & 16) sP[r * SEQ + limit + sub] = 0;
        __syncthreads();
        if (tid < 16) {
            float s = 0.f;
            for (int j = 0; j < 16; ++j) s += sRed[tid * 16 + j];
            sInv[tid] = 1.0f / s;
        }
        __syncthreads();
    }

    // ---- O = P @ V : wave owns d_v slice [wave*128, wave*128+128) ----
    const int n_base = wave * 128;
    const int limit2 = (limit + 31) & ~31;
    v8f oacc[8];
#pragma unroll
    for (int ni = 0; ni < 8; ++ni)
        oacc[ni] = (v8f){0.f, 0.f, 0.f, 0.f, 0.f, 0.f, 0.f, 0.f};

    const unsigned short* vcol =
        Vt + ((size_t)b * DMODEL + n_base + l15) * SEQ + lhalf * 16;

#pragma unroll 1
    for (int k0 = 0; k0 < limit2; k0 += 32) {
        FragBF a;
        const unsigned short* pa = sP + l15 * SEQ + k0 + lhalf * 16;
        a.q[0] = *(const uint4*)pa;
        a.q[1] = *(const uint4*)(pa + 8);
#pragma unroll
        for (int ni = 0; ni < 8; ++ni) {
            FragBF bv;
            const unsigned short* pb = vcol + (size_t)ni * 16 * SEQ + k0;
            bv.q[0] = *(const uint4*)pb;
            bv.q[1] = *(const uint4*)(pb + 8);
            oacc[ni] = wmma_bf16(a, bv, oacc[ni]);
        }
    }

#pragma unroll
    for (int ni = 0; ni < 8; ++ni)
#pragma unroll
        for (int i = 0; i < 8; ++i) {
            const int r   = i + lhalf * 8;
            const int col = n_base + ni * 16 + l15;
            const float v = oacc[ni][i] * sInv[r];
            Ob[((size_t)(b * SEQ + q0 + r)) * DMODEL + col] = f32_to_bf16(v);
        }
}

// ---------------------------------------------------------------------------
// Host-side orchestration
// ---------------------------------------------------------------------------
extern "C" void kernel_launch(void* const* d_in, const int* in_sizes, int n_in,
                              void* d_out, int out_size, void* d_ws, size_t ws_size,
                              hipStream_t stream) {
    const float* x  = (const float*)d_in[0];
    const float* Wq = (const float*)d_in[1];
    const float* Wk = (const float*)d_in[2];
    const float* Wv = (const float*)d_in[3];
    const float* Wo = (const float*)d_in[4];
    float* out = (float*)d_out;

    char* ws = (char*)d_ws;
    size_t off = 0;
    auto carve = [&](size_t bytes) -> void* {
        void* p = ws + off;
        off += (bytes + 255) & ~(size_t)255;
        return p;
    };
    unsigned short* xb  = (unsigned short*)carve((size_t)MROWS * DMODEL * 2);
    unsigned short* wqb = (unsigned short*)carve((size_t)DMODEL * DMODEL * 2);
    unsigned short* wkb = (unsigned short*)carve((size_t)DMODEL * DMODEL * 2);
    unsigned short* wvb = (unsigned short*)carve((size_t)DMODEL * DMODEL * 2);
    unsigned short* wob = (unsigned short*)carve((size_t)DMODEL * DMODEL * 2);
    unsigned short* Qb  = (unsigned short*)carve((size_t)MROWS * DMODEL * 2);
    unsigned short* Kb  = (unsigned short*)carve((size_t)MROWS * DMODEL * 2);
    unsigned short* Vt  = (unsigned short*)carve((size_t)MROWS * DMODEL * 2);  // [B, D, S]
    unsigned short* Ob  = (unsigned short*)carve((size_t)MROWS * DMODEL * 2);

    // fp32 -> bf16
    cvt_f32_to_bf16<<<4096, 256, 0, stream>>>(x,  xb,  MROWS * DMODEL);
    cvt_f32_to_bf16<<<1024, 256, 0, stream>>>(Wq, wqb, DMODEL * DMODEL);
    cvt_f32_to_bf16<<<1024, 256, 0, stream>>>(Wk, wkb, DMODEL * DMODEL);
    cvt_f32_to_bf16<<<1024, 256, 0, stream>>>(Wv, wvb, DMODEL * DMODEL);
    cvt_f32_to_bf16<<<1024, 256, 0, stream>>>(Wo, wob, DMODEL * DMODEL);

    // projections: block tile 256(M) x 128(N), wave tile 64x64
    dim3 gg(DMODEL / 128, MROWS / 256);   // (8, 64)
    gemm_bf16_wmma<0><<<gg, 256, 0, stream>>>(xb, wqb, Qb);
    gemm_bf16_wmma<0><<<gg, 256, 0, stream>>>(xb, wkb, Kb);
    gemm_bf16_wmma<1><<<gg, 256, 0, stream>>>(xb, wvb, Vt);   // transposed store

    // attention
    const size_t smem = (size_t)16 * SEQ * 4      // scores fp32
                      + (size_t)16 * SEQ * 2      // probs bf16
                      + (size_t)16 * DMODEL * 2   // Q tile bf16
                      + (256 + 16 + 16) * 4;      // reductions
    flash_attn<<<dim3(SEQ / 16, BATCH), 256, smem, stream>>>(Qb, Kb, Vt, Ob);

    // output projection -> fp32 d_out
    gemm_bf16_wmma<2><<<gg, 256, 0, stream>>>(Ob, wob, out);
}
